// SAGEEncoder_27565100106034
// MI455X (gfx1250) — matrix operand order repaired and checked by reading, not compiled, not run on hardware
//
#include <hip/hip_runtime.h>

typedef __bf16 bf16_t;
typedef __attribute__((ext_vector_type(16))) __bf16        v16bf;
typedef __attribute__((ext_vector_type(8)))  float         v8f;
typedef __attribute__((ext_vector_type(4)))  unsigned int  u32x4;

#define NIN    300
#define K0PAD  320     // 300 padded to multiple of 32
#define HIDF   256
#define NOUTF  768
#define NGRAPH 256

static inline unsigned cdiv(size_t a, unsigned b) { return (unsigned)((a + b - 1) / b); }

__device__ __forceinline__ bf16_t f2bf(float x) {
    unsigned u = __builtin_bit_cast(unsigned, x);
    unsigned r = u + 0x7FFFu + ((u >> 16) & 1u);   // round-to-nearest-even
    unsigned short h = (unsigned short)(r >> 16);
    return __builtin_bit_cast(bf16_t, h);
}

// ---------------- degree / scatter (mean aggregation) ----------------

__global__ void k_deg(const long long* __restrict__ dst, float* __restrict__ deg, int nE) {
    int e = blockIdx.x * blockDim.x + threadIdx.x;
    if (e < nE) unsafeAtomicAdd(&deg[(int)dst[e]], 1.0f);
}

__global__ void k_scatter(const float* __restrict__ h, const long long* __restrict__ src,
                          const long long* __restrict__ dst, float* __restrict__ sum, int F) {
    int e = blockIdx.x;
    int s = (int)src[e];
    int d = (int)dst[e];
    const float* hs = h + (size_t)s * F;
    float*       sd = sum + (size_t)d * F;
    for (int f = threadIdx.x; f < F; f += blockDim.x)
        unsafeAtomicAdd(&sd[f], hs[f]);
}

// ---------------- conversions to padded bf16 ----------------

// out[r][c] = (c<Fin) ? bf16(in[r][c]) : 0
__global__ void k_cvt_pad(const float* __restrict__ in, bf16_t* __restrict__ out,
                          int rows, int Fin, int Kpad) {
    size_t i = (size_t)blockIdx.x * blockDim.x + threadIdx.x;
    size_t total = (size_t)rows * Kpad;
    if (i >= total) return;
    int r = (int)(i / Kpad), c = (int)(i % Kpad);
    float v = (c < Fin) ? in[(size_t)r * Fin + c] : 0.0f;
    out[i] = f2bf(v);
}

// out[r][c] = (c<Fin) ? bf16(sum[r][c] / max(cnt[r],1)) : 0   (also used for pooling)
__global__ void k_cvt_mean(const float* __restrict__ sum, const float* __restrict__ cnt,
                           bf16_t* __restrict__ out, int rows, int Fin, int Kpad) {
    size_t i = (size_t)blockIdx.x * blockDim.x + threadIdx.x;
    size_t total = (size_t)rows * Kpad;
    if (i >= total) return;
    int r = (int)(i / Kpad), c = (int)(i % Kpad);
    float v = 0.0f;
    if (c < Fin) v = sum[(size_t)r * Fin + c] / fmaxf(cnt[r], 1.0f);
    out[i] = f2bf(v);
}

// W is [din][dout] f32 row-major; Wt is [dout][Kpad] bf16 (transposed, zero-padded K)
__global__ void k_cvt_w(const float* __restrict__ W, bf16_t* __restrict__ Wt,
                        int din, int dout, int Kpad) {
    size_t i = (size_t)blockIdx.x * blockDim.x + threadIdx.x;
    size_t total = (size_t)dout * Kpad;
    if (i >= total) return;
    int o = (int)(i / Kpad), c = (int)(i % Kpad);
    float v = (c < din) ? W[(size_t)c * dout + o] : 0.0f;
    Wt[i] = f2bf(v);
}

// ---------------- pooling ----------------

__global__ void k_pool_cnt(const long long* __restrict__ batch, float* __restrict__ cnt, int n) {
    int i = blockIdx.x * blockDim.x + threadIdx.x;
    if (i < n) unsafeAtomicAdd(&cnt[(int)batch[i]], 1.0f);
}

__global__ void k_pool_sum(const float* __restrict__ h, const long long* __restrict__ batch,
                           float* __restrict__ sum, int F) {
    int nidx = blockIdx.x;
    int g = (int)batch[nidx];
    const float* hn = h + (size_t)nidx * F;
    float*       sg = sum + (size_t)g * F;
    for (int f = threadIdx.x; f < F; f += blockDim.x)
        unsafeAtomicAdd(&sg[f], hn[f]);
}

// ---------------- WMMA GEMM ----------------
// Fragment loads match CDNA5 ISA 16-bit A (16x32) / B (32x16) VGPR layouts.

union Frag32B { u32x4 u[2]; v16bf v; };

__device__ __forceinline__ v16bf load_a_frag(const bf16_t* __restrict__ A, int row,
                                             int Kpad, int k0, int lane) {
    // lane 0-15: K = k0+0..7 (v0..3) and k0+16..23 (v4..7); lane 16-31: +8
    int bk = (lane >> 4) * 8;
    const bf16_t* p = A + (size_t)row * Kpad + k0 + bk;
    Frag32B f;
    f.u[0] = *reinterpret_cast<const u32x4*>(p);        // K = k0+bk .. +7
    f.u[1] = *reinterpret_cast<const u32x4*>(p + 16);   // K = k0+bk+16 .. +23
    return f.v;
}

__device__ __forceinline__ v16bf load_b_frag(const bf16_t* __restrict__ Bt, int col,
                                             int Kpad, int k0, int lane) {
    // lane 0-15: K = k0+0..7 and k0+16..23 for column n=lane; lane 16-31: +8
    int bk = (lane >> 4) * 8;
    const bf16_t* p = Bt + (size_t)col * Kpad + k0 + bk;
    Frag32B f;
    f.u[0] = *reinterpret_cast<const u32x4*>(p);
    f.u[1] = *reinterpret_cast<const u32x4*>(p + 16);
    return f.v;
}

// out[M,N] = act( A0@B0t^T + (DUAL ? A1@B1t^T : 0) + bias )
// One wave computes a 16x64 strip: 1 M-tile x 4 N-tiles. Per K-step all fragment
// loads for a group are issued before its 4 WMMAs so loads overlap matrix ops.
// Block = 8 waves = 8 M-tiles x 4 N-tiles (128x64 macro-tile).
template <bool DUAL>
__global__ void k_gemm(const bf16_t* __restrict__ A0, const bf16_t* __restrict__ B0t,
                       const bf16_t* __restrict__ A1, const bf16_t* __restrict__ B1t,
                       const float* __restrict__ bias, float* __restrict__ out,
                       int Mrows, int Kpad, int ldout, int doRelu) {
    int lane   = threadIdx.x & 31;
    int w      = threadIdx.x >> 5;             // 8 waves per block
    int mt     = blockIdx.x * 8 + w;           // one M-tile per wave
    int ntBase = blockIdx.y * 4;               // 4 N-tiles per wave
    int Mtiles = (Mrows + 15) >> 4;
    if (mt >= Mtiles) return;                  // whole-wave exit: EXEC stays all-1s

    int rrow = mt * 16 + (lane & 15);
    if (rrow >= Mrows) rrow = Mrows - 1;       // clamp OOB gather (tiles are exact here)
    int col0 = ntBase * 16 + (lane & 15);      // column for N-tile j is col0 + 16*j

    v8f acc[4];
#pragma unroll
    for (int j = 0; j < 4; ++j) acc[j] = v8f{0.f, 0.f, 0.f, 0.f, 0.f, 0.f, 0.f, 0.f};

    for (int k0 = 0; k0 < Kpad; k0 += 32) {
        // group 0: issue all loads, then 4 WMMAs
        v16bf a0 = load_a_frag(A0, rrow, Kpad, k0, lane);
        v16bf b0[4];
#pragma unroll
        for (int j = 0; j < 4; ++j) b0[j] = load_b_frag(B0t, col0 + 16 * j, Kpad, k0, lane);
#pragma unroll
        for (int j = 0; j < 4; ++j)
            acc[j] = __builtin_amdgcn_wmma_f32_16x16x32_bf16(false, a0, false, b0[j],
                                                             (short)0, acc[j], false, false);
        if constexpr (DUAL) {
            // group 1 loads are independent of group 0 WMMAs -> can issue under them
            v16bf a1 = load_a_frag(A1, rrow, Kpad, k0, lane);
            v16bf b1[4];
#pragma unroll
            for (int j = 0; j < 4; ++j) b1[j] = load_b_frag(B1t, col0 + 16 * j, Kpad, k0, lane);
#pragma unroll
            for (int j = 0; j < 4; ++j)
                acc[j] = __builtin_amdgcn_wmma_f32_16x16x32_bf16(false, a1, false, b1[j],
                                                                 (short)0, acc[j], false, false);
        }
    }

#pragma unroll
    for (int j = 0; j < 4; ++j) {
        int col = col0 + 16 * j;
        float bcol = bias ? bias[col] : 0.0f;
#pragma unroll
        for (int r = 0; r < 8; ++r) {
            int orow = mt * 16 + r + ((lane >> 4) << 3);  // C/D layout: v_r -> M=r (+8 hi lanes)
            if (orow < Mrows) {
                float v = acc[j][r] + bcol;
                if (doRelu) v = fmaxf(v, 0.0f);
                out[(size_t)orow * ldout + col] = v;
            }
        }
    }
}

// ---------------- host side ----------------

extern "C" void kernel_launch(void* const* d_in, const int* in_sizes, int n_in,
                              void* d_out, int out_size, void* d_ws, size_t ws_size,
                              hipStream_t stream) {
    const float*     x     = (const float*)d_in[0];
    const long long* ei    = (const long long*)d_in[1];
    const long long* batch = (const long long*)d_in[2];
    const float* Wl[4] = {(const float*)d_in[3], (const float*)d_in[6],
                          (const float*)d_in[9], (const float*)d_in[12]};
    const float* bb[4] = {(const float*)d_in[4], (const float*)d_in[7],
                          (const float*)d_in[10], (const float*)d_in[13]};
    const float* Wr[4] = {(const float*)d_in[5], (const float*)d_in[8],
                          (const float*)d_in[11], (const float*)d_in[14]};
    const float* fc1w = (const float*)d_in[15];
    const float* fc1b = (const float*)d_in[16];
    const float* fc2w = (const float*)d_in[17];
    const float* fc2b = (const float*)d_in[18];

    const int N  = in_sizes[0] / NIN;      // 50000
    const int nE = in_sizes[1] / 2;        // 800000
    const long long* src  = ei;
    const long long* dstE = ei + nE;

    // workspace carve-out (256B aligned)
    char* p = (char*)d_ws;
    auto carve = [&](size_t bytes) -> char* {
        char* r = p;
        p += (bytes + 255) & ~(size_t)255;
        return r;
    };
    float*  hA   = (float*)carve((size_t)N * HIDF * 4);
    float*  hB   = (float*)carve((size_t)N * HIDF * 4);
    float*  agg  = (float*)carve((size_t)N * NIN * 4);
    float*  deg  = (float*)carve((size_t)N * 4);
    bf16_t* abf  = (bf16_t*)carve((size_t)N * K0PAD * 2);
    bf16_t* hbf  = (bf16_t*)carve((size_t)N * K0PAD * 2);
    bf16_t* wlt[4], *wrt[4];
    for (int i = 0; i < 4; ++i) {
        int kp = (i == 0) ? K0PAD : HIDF;
        wlt[i] = (bf16_t*)carve((size_t)HIDF * kp * 2);
        wrt[i] = (bf16_t*)carve((size_t)HIDF * kp * 2);
    }
    bf16_t* fc1wt = (bf16_t*)carve((size_t)HIDF * HIDF * 2);
    bf16_t* fc2wt = (bf16_t*)carve((size_t)NOUTF * HIDF * 2);
    float*  psum  = (float*)carve((size_t)NGRAPH * HIDF * 4);
    float*  pcnt  = (float*)carve((size_t)NGRAPH * 4);
    bf16_t* pbf   = (bf16_t*)carve((size_t)NGRAPH * HIDF * 2);
    float*  f1o   = (float*)carve((size_t)NGRAPH * HIDF * 4);
    bf16_t* f1bf  = (bf16_t*)carve((size_t)NGRAPH * HIDF * 2);
    (void)ws_size; (void)n_in; (void)out_size;

    // weights -> transposed, padded bf16
    for (int i = 0; i < 4; ++i) {
        int din = (i == 0) ? NIN : HIDF;
        int kp  = (i == 0) ? K0PAD : HIDF;
        size_t tot = (size_t)HIDF * kp;
        k_cvt_w<<<cdiv(tot, 256), 256, 0, stream>>>(Wl[i], wlt[i], din, HIDF, kp);
        k_cvt_w<<<cdiv(tot, 256), 256, 0, stream>>>(Wr[i], wrt[i], din, HIDF, kp);
    }
    k_cvt_w<<<cdiv((size_t)HIDF * HIDF, 256), 256, 0, stream>>>(fc1w, fc1wt, HIDF, HIDF, HIDF);
    k_cvt_w<<<cdiv((size_t)NOUTF * HIDF, 256), 256, 0, stream>>>(fc2w, fc2wt, HIDF, NOUTF, HIDF);

    // degrees (graph topology fixed across layers)
    hipMemsetAsync(deg, 0, (size_t)N * 4, stream);
    k_deg<<<cdiv((size_t)nE, 256), 256, 0, stream>>>(dstE, deg, nE);

    // SAGE layers
    const float* hin = x;
    float* houts[4] = {hA, hB, hA, hB};
    const int Mtiles = (N + 15) / 16;
    dim3 ggrid(cdiv((size_t)Mtiles, 8), (HIDF / 16) / 4);   // 8 M-tiles, 4 N-tiles per block
    for (int i = 0; i < 4; ++i) {
        int din = (i == 0) ? NIN : HIDF;
        int kp  = (i == 0) ? K0PAD : HIDF;
        hipMemsetAsync(agg, 0, (size_t)N * din * 4, stream);
        k_scatter<<<nE, 256, 0, stream>>>(hin, src, dstE, agg, din);
        size_t tot = (size_t)N * kp;
        k_cvt_mean<<<cdiv(tot, 256), 256, 0, stream>>>(agg, deg, abf, N, din, kp);
        k_cvt_pad <<<cdiv(tot, 256), 256, 0, stream>>>(hin, hbf, N, din, kp);
        // relu after every layer (3 inter-layer + final explicit relu)
        k_gemm<true><<<ggrid, 256, 0, stream>>>(abf, wlt[i], hbf, wrt[i], bb[i],
                                                houts[i], N, kp, HIDF, 1);
        hin = houts[i];
    }

    // global mean pool
    hipMemsetAsync(psum, 0, (size_t)NGRAPH * HIDF * 4, stream);
    hipMemsetAsync(pcnt, 0, (size_t)NGRAPH * 4, stream);
    k_pool_cnt<<<cdiv((size_t)N, 256), 256, 0, stream>>>(batch, pcnt, N);
    k_pool_sum<<<N, 256, 0, stream>>>(hin, batch, psum, HIDF);
    k_cvt_mean<<<cdiv((size_t)NGRAPH * HIDF, 256), 256, 0, stream>>>(psum, pcnt, pbf,
                                                                     NGRAPH, HIDF, HIDF);

    // fc1 (relu) then fc2 (bias only) -> d_out [256 x 768] f32
    dim3 g1(2, 4);    // 16 M-tiles / 8, 16 N-tiles / 4
    k_gemm<false><<<g1, 256, 0, stream>>>(pbf, fc1wt, nullptr, nullptr, fc1b,
                                          f1o, NGRAPH, HIDF, HIDF, 1);
    k_cvt_pad<<<cdiv((size_t)NGRAPH * HIDF, 256), 256, 0, stream>>>(f1o, f1bf,
                                                                    NGRAPH, HIDF, HIDF);
    dim3 g2(2, 12);   // 16 M-tiles / 8, 48 N-tiles / 4
    k_gemm<false><<<g2, 256, 0, stream>>>(f1bf, fc2wt, nullptr, nullptr, fc2b,
                                          (float*)d_out, NGRAPH, HIDF, NOUTF, 0);
}
